// MultiGeometryAttention_9964324127129
// MI455X (gfx1250) — matrix operand order repaired and checked by reading, not compiled, not run on hardware
//
#include <hip/hip_runtime.h>
#include <hip/hip_bf16.h>
#include <math.h>

// ---------------- problem constants (match reference) ----------------
#define Bn 4
#define Tn 2048
#define Cn 1024
#define Hn 16
#define Dn 64
#define BHTD (Bn * Hn * Tn * Dn) // 8388608 elems per q/k/v plane

typedef __bf16 v16bf __attribute__((ext_vector_type(16)));
typedef float  v8f   __attribute__((ext_vector_type(8)));
typedef unsigned int tv4u __attribute__((ext_vector_type(4)));
typedef int          tv8i __attribute__((ext_vector_type(8)));
typedef int          tv4i __attribute__((ext_vector_type(4)));

// native f32 -> bf16 (backend emits hardware cvt on gfx1250)
__device__ __forceinline__ __bf16 f2bf(float f) { return (__bf16)f; }

// LDS byte offset of a __shared__ object (generic shared addr low 32 bits)
__device__ __forceinline__ unsigned lds_off(const void* p) {
    return (unsigned)(unsigned long long)(uintptr_t)p;
}

// ---------------- Tensor Data Mover helpers (ISA ch.8 D#) ------------
// Group1 for a 2D tile of bf16: tile_d0 contiguous elems per row, tile_d1 rows,
// row stride stride0 elems. LDS padding: +4 dwords after every 16 dwords
// => rows land at stride 40 bf16 (LDT), matching the fragment loaders.
__device__ __forceinline__ tv8i tdm_g1_2d(unsigned tile_d0, unsigned tile_d1,
                                          unsigned long long stride0) {
    tv8i g1;
    g1[0] = (int)((1u << 16)        // data_size = 1 (2 bytes)
                | (1u << 20)        // pad_enable
                | (3u << 22)        // pad_interval: 16 dwords (one 32-bf16 row)
                | (3u << 25));      // pad_amount: 4 dwords (8 bf16)
    g1[1] = (int)((tile_d0 & 0xFFFFu) << 16);                      // tensor_dim0 lo
    g1[2] = (int)((tile_d0 >> 16) | ((tile_d1 & 0xFFFFu) << 16));  // dim0 hi | dim1 lo
    g1[3] = (int)((tile_d1 >> 16) | (tile_d0 << 16));              // dim1 hi | tile_dim0
    g1[4] = (int)(tile_d1 & 0xFFFFu);                              // tile_dim1 (tile_dim2=0)
    g1[5] = (int)(stride0 & 0xFFFFFFFFull);                        // tensor_dim0_stride lo
    g1[6] = (int)((stride0 >> 32) & 0xFFFFull);                    // stride hi (dim1_stride=0)
    g1[7] = 0;
    return g1;
}

__device__ __forceinline__ void tdm_load_2d(unsigned lds_byte_off,
                                            const __bf16* gptr, tv8i g1) {
    unsigned long long ga = (unsigned long long)(uintptr_t)gptr;
    tv4u g0;
    g0[0] = 1u;                                             // count=1 (user D#)
    g0[1] = lds_byte_off;                                   // lds_addr
    g0[2] = (unsigned)ga;                                   // global_addr[31:0]
    g0[3] = (unsigned)((ga >> 32) & 0x01FFFFFFu) | (2u << 30); // addr[56:32] | type=2
    tv4i z4 = {0, 0, 0, 0};
    tv8i z8 = {0, 0, 0, 0, 0, 0, 0, 0};
    // 6-arg form (clang-23 / therock headers): (g0, g1, g2, g3, extra, cpol)
    __builtin_amdgcn_tensor_load_to_lds(g0, g1, z4, z4, z8, 0);
}

// ---------------- WMMA fragment loaders (ISA 7.12.2 layouts) ---------
__device__ __forceinline__ v16bf load_a(const __bf16* base, int ld) {
    const int lane = threadIdx.x & 31;
    const int hh = lane >> 4, r = lane & 15;
    const __bf16* p = base + r * ld;
    v16bf a;
#pragma unroll
    for (int j = 0; j < 16; ++j) {
        int kk = j + hh * 8 + ((j >= 8) ? 8 : 0);
        a[j] = p[kk];
    }
    return a;
}
__device__ __forceinline__ v16bf load_b(const __bf16* base, int ld) {
    const int lane = threadIdx.x & 31;
    const int hh = lane >> 4, c = lane & 15;
    const __bf16* p = base + c * ld;
    v16bf b;
#pragma unroll
    for (int j = 0; j < 16; ++j) b[j] = p[j + hh * 8 + ((j >= 8) ? 8 : 0)];
    return b;
}

#define LDT 40 // padded bf16 row stride written by TDM (32 data + 8 pad)

// ============== Kernel 0: f32 -> bf16 pre-convert ====================
__global__ __launch_bounds__(256) void conv_bf16(const float* __restrict__ in,
                                                 __bf16* __restrict__ out) {
    int i = (blockIdx.x * 256 + threadIdx.x) * 4;
    float4 f = *(const float4*)(in + i);
    out[i + 0] = f2bf(f.x); out[i + 1] = f2bf(f.y);
    out[i + 2] = f2bf(f.z); out[i + 3] = f2bf(f.w);
}

// ================= Kernel 1: QKV projection GEMM =====================
// qkv[m][n] = sum_k x[m][k]*Wqkv[n][k]; M=8192, N=3072, K=1024.
// TDM double-buffered bf16 tiles; epilogue scatters into [B][H][T][D] f32.
__global__ __launch_bounds__(256) void qkv_gemm(const __bf16* __restrict__ Xb,
                                                const __bf16* __restrict__ Wb,
                                                float* __restrict__ qkv) {
    __shared__ __bf16 As[2][128][LDT];
    __shared__ __bf16 Bs[2][128][LDT];
    const int tid = threadIdx.x;
    const int m0 = blockIdx.x * 128;
    const int n0 = blockIdx.y * 128;
    const int w = tid >> 5, lane = tid & 31;
    const int wm = (w & 3) * 32, wn = (w >> 2) * 64;

    const tv8i g1 = tdm_g1_2d(32u, 128u, 1024ull);
    v8f acc[2][4] = {};

    if (w == 0) {
        tdm_load_2d(lds_off(&As[0][0][0]), Xb + (size_t)m0 * Cn, g1);
        tdm_load_2d(lds_off(&Bs[0][0][0]), Wb + (size_t)n0 * Cn, g1);
    }
    for (int kt = 0; kt < 32; ++kt) {
        const int cur = kt & 1, nxt = cur ^ 1;
        __syncthreads(); // all waves done reading buffer `nxt` (from kt-1)
        if (w == 0) {
            if (kt + 1 < 32) {
                int k1 = (kt + 1) * 32;
                tdm_load_2d(lds_off(&As[nxt][0][0]), Xb + (size_t)m0 * Cn + k1, g1);
                tdm_load_2d(lds_off(&Bs[nxt][0][0]), Wb + (size_t)n0 * Cn + k1, g1);
                __builtin_amdgcn_s_wait_tensorcnt(2); // stage kt retired
            } else {
                __builtin_amdgcn_s_wait_tensorcnt(0);
            }
        }
        __syncthreads(); // stage kt visible to all waves
        v16bf a0 = load_a(&As[cur][wm][0], LDT);
        v16bf a1 = load_a(&As[cur][wm + 16][0], LDT);
        v16bf b0 = load_b(&Bs[cur][wn][0], LDT);
        v16bf b1 = load_b(&Bs[cur][wn + 16][0], LDT);
        v16bf b2 = load_b(&Bs[cur][wn + 32][0], LDT);
        v16bf b3 = load_b(&Bs[cur][wn + 48][0], LDT);
        acc[0][0] = __builtin_amdgcn_wmma_f32_16x16x32_bf16(false, a0, false, b0, (short)0, acc[0][0], false, false);
        acc[0][1] = __builtin_amdgcn_wmma_f32_16x16x32_bf16(false, a0, false, b1, (short)0, acc[0][1], false, false);
        acc[0][2] = __builtin_amdgcn_wmma_f32_16x16x32_bf16(false, a0, false, b2, (short)0, acc[0][2], false, false);
        acc[0][3] = __builtin_amdgcn_wmma_f32_16x16x32_bf16(false, a0, false, b3, (short)0, acc[0][3], false, false);
        acc[1][0] = __builtin_amdgcn_wmma_f32_16x16x32_bf16(false, a1, false, b0, (short)0, acc[1][0], false, false);
        acc[1][1] = __builtin_amdgcn_wmma_f32_16x16x32_bf16(false, a1, false, b1, (short)0, acc[1][1], false, false);
        acc[1][2] = __builtin_amdgcn_wmma_f32_16x16x32_bf16(false, a1, false, b2, (short)0, acc[1][2], false, false);
        acc[1][3] = __builtin_amdgcn_wmma_f32_16x16x32_bf16(false, a1, false, b3, (short)0, acc[1][3], false, false);
    }
    const int hh = lane >> 4, c = lane & 15;
#pragma unroll
    for (int mi = 0; mi < 2; ++mi) {
#pragma unroll
        for (int s = 0; s < 4; ++s) {
            int n = n0 + wn + s * 16 + c;
            int part = n >> 10, cc = n & 1023;
            int hd = cc >> 6, d = cc & 63;
            float* base = qkv + (size_t)part * BHTD;
#pragma unroll
            for (int i = 0; i < 8; ++i) {
                int m = m0 + wm + mi * 16 + i + 8 * hh;
                int bb = m >> 11, t = m & (Tn - 1);
                base[(((size_t)(bb * Hn + hd)) * Tn + t) * Dn + d] = acc[mi][s][i];
            }
        }
    }
}

// ====================== Kernel 2: RoPE (q and k) =====================
__global__ __launch_bounds__(256) void rope_kernel(float* __restrict__ q,
                                                   float* __restrict__ k) {
    int idx = blockIdx.x * 256 + threadIdx.x; // B*H*T*32 threads
    int i = idx & 31;
    int row = idx >> 5;
    int t = row & (Tn - 1);
    float inv = __powf(10000.0f, -(float)i / 32.0f);
    float ang = (float)t * inv;
    float cs = __cosf(ang), sn = __sinf(ang);
    size_t base = (size_t)row * Dn;
    float a1 = q[base + i], a2 = q[base + 32 + i];
    q[base + i] = a1 * cs + a2 * sn;
    q[base + 32 + i] = -a1 * sn + a2 * cs;
    float b1 = k[base + i], b2 = k[base + 32 + i];
    k[base + i] = b1 * cs + b2 * sn;
    k[base + 32 + i] = -b1 * sn + b2 * cs;
}

// ============ Kernel 3: 3-geometry causal flash attention ============
// grid (T/64, H, B), 128 threads = 4 waves; wave w owns 16 q rows.
// Output written as bf16 to attb [B][H][T][D] (consumed by TDM in out_proj).
__global__ __launch_bounds__(128) void attn_kernel(const float* __restrict__ q,
                                                   const float* __restrict__ k,
                                                   const float* __restrict__ v,
                                                   const float* __restrict__ x0w,
                                                   const float* __restrict__ x0b,
                                                   const float* __restrict__ hcur,
                                                   __bf16* __restrict__ attb) {
    __shared__ __bf16 Qs[64][72];
    __shared__ __bf16 Ks[32][72];
    __shared__ __bf16 Vt[64][40];     // V transposed: [d][key]
    __shared__ __bf16 Ps[4][16][40];  // per-wave P tiles
    __shared__ float tkS[32];
    __shared__ float tqS[64];

    const int qb = blockIdx.x, h = blockIdx.y, b = blockIdx.z;
    const int tid = threadIdx.x, w = tid >> 5, lane = tid & 31;
    const int variant = (h < 6) ? 0 : (h < 10 ? 1 : 2);
    const float kc = (variant == 1) ? hcur[h - 6] : 1.0f;
    const float rsq_kc = rsqrtf(kc);
    const float x0bv = x0b[0];
    const size_t headBase = ((size_t)(b * Hn + h)) * Tn * Dn;

    { // ---- Q stage: 2 lanes per row, 32 elems each ----
        const int r = lane & 15, dh = lane >> 4;
        const int qrow = qb * 64 + w * 16 + r;
        const float* qp = q + headBase + (size_t)qrow * Dn + dh * 32;
        float qv[32];
        float ss = 0.f, dt = 0.f;
#pragma unroll
        for (int j = 0; j < 32; j += 4) {
            float4 f = *(const float4*)(qp + j);
            qv[j] = f.x; qv[j + 1] = f.y; qv[j + 2] = f.z; qv[j + 3] = f.w;
        }
#pragma unroll
        for (int j = 0; j < 32; ++j) {
            ss += qv[j] * qv[j];
            if (variant == 1) dt += qv[j] * x0w[dh * 32 + j];
        }
        ss += __shfl_xor(ss, 16, 32);
        dt += __shfl_xor(dt, 16, 32);
        float rn = 1.0f / fmaxf(sqrtf(ss), 1e-12f);
        float mul, ssa = 0.f;
        if (variant == 0) mul = 0.125f;           // 1/sqrt(D)
        else if (variant == 2) mul = 8.0f * rn;   // sqrt(D) * l2norm
        else { float qn = dt + x0bv; mul = 1.0f + qn * rn; ssa = mul * mul * ss; }
#pragma unroll
        for (int j = 0; j < 32; ++j) Qs[w * 16 + r][dh * 32 + j] = f2bf(qv[j] * mul);
        if (variant == 1 && dh == 0) tqS[w * 16 + r] = sqrtf(1.0f / kc + ssa);
    }
    v16bf aq_lo = load_a(&Qs[w * 16][0], 72);  // wave-local DS RAW
    v16bf aq_hi = load_a(&Qs[w * 16][32], 72);

    float mrow[8], lrow[8];
#pragma unroll
    for (int i = 0; i < 8; ++i) { mrow[i] = -3.0e38f; lrow[i] = 0.f; }
    v8f oacc[4] = {};

    const int hh = lane >> 4, c = lane & 15;
    const int nkb = 2 * qb + 2; // causal bound
    for (int kb = 0; kb < nkb; ++kb) {
        __syncthreads();
        { // cooperative K/V load+transform: 4 threads per key row
            const int row = tid >> 2, qd = tid & 3, dbase = qd * 16;
            const int kk = kb * 32 + row;
            const float* kp = k + headBase + (size_t)kk * Dn + dbase;
            const float* vp = v + headBase + (size_t)kk * Dn + dbase;
            float kg[16], vg[16];
#pragma unroll
            for (int j = 0; j < 16; j += 4) {
                float4 f = *(const float4*)(kp + j);
                kg[j] = f.x; kg[j + 1] = f.y; kg[j + 2] = f.z; kg[j + 3] = f.w;
                float4 g = *(const float4*)(vp + j);
                vg[j] = g.x; vg[j + 1] = g.y; vg[j + 2] = g.z; vg[j + 3] = g.w;
            }
            float ss = 0.f, dt = 0.f;
#pragma unroll
            for (int j = 0; j < 16; ++j) {
                ss += kg[j] * kg[j];
                if (variant == 1) dt += kg[j] * x0w[dbase + j];
            }
            ss += __shfl_xor(ss, 1, 32); ss += __shfl_xor(ss, 2, 32);
            dt += __shfl_xor(dt, 1, 32); dt += __shfl_xor(dt, 2, 32);
            float rn = 1.0f / fmaxf(sqrtf(ss), 1e-12f);
            float mul = 1.0f, ssa = 0.f;
            if (variant == 2) mul = rn;
            else if (variant == 1) { float kn = dt + x0bv; mul = 1.0f + kn * rn; ssa = mul * mul * ss; }
#pragma unroll
            for (int j = 0; j < 16; ++j) {
                Ks[row][dbase + j] = f2bf(kg[j] * mul);
                Vt[dbase + j][row] = f2bf(vg[j]);
            }
            if (variant == 1 && qd == 0) tkS[row] = sqrtf(1.0f / kc + ssa);
        }
        __syncthreads();

        v8f sacc[2];
#pragma unroll
        for (int s = 0; s < 2; ++s) {
            v8f z = {};
            v16bf blo = load_b(&Ks[s * 16][0], 72);
            v16bf bhi = load_b(&Ks[s * 16][32], 72);
            z = __builtin_amdgcn_wmma_f32_16x16x32_bf16(false, aq_lo, false, blo, (short)0, z, false, false);
            z = __builtin_amdgcn_wmma_f32_16x16x32_bf16(false, aq_hi, false, bhi, (short)0, z, false, false);
            sacc[s] = z;
        }
#pragma unroll
        for (int i = 0; i < 8; ++i) {
            const int m_local = i + 8 * hh;
            const int qq = qb * 64 + w * 16 + m_local;
            float l0 = sacc[0][i], l1 = sacc[1][i];
            if (variant == 1) {
                float tq = tqS[w * 16 + m_local];
                float a0 = fmaxf(-kc * (l0 - tq * tkS[c]),      1.0f + 1e-7f);
                float a1 = fmaxf(-kc * (l1 - tq * tkS[16 + c]), 1.0f + 1e-7f);
                l0 = -acoshf(a0) * rsq_kc;
                l1 = -acoshf(a1) * rsq_kc;
            }
            if (kb * 32 + c > qq)      l0 = -1.0e30f;
            if (kb * 32 + 16 + c > qq) l1 = -1.0e30f;
            float rmax = fmaxf(l0, l1);
            rmax = fmaxf(rmax, __shfl_xor(rmax, 1, 32));
            rmax = fmaxf(rmax, __shfl_xor(rmax, 2, 32));
            rmax = fmaxf(rmax, __shfl_xor(rmax, 4, 32));
            rmax = fmaxf(rmax, __shfl_xor(rmax, 8, 32));
            float mnew = fmaxf(mrow[i], rmax);
            float scale = __expf(mrow[i] - mnew);
            float p0 = __expf(l0 - mnew), p1 = __expf(l1 - mnew);
            float rs = p0 + p1;
            rs += __shfl_xor(rs, 1, 32); rs += __shfl_xor(rs, 2, 32);
            rs += __shfl_xor(rs, 4, 32); rs += __shfl_xor(rs, 8, 32);
            lrow[i] = lrow[i] * scale + rs;
            mrow[i] = mnew;
#pragma unroll
            for (int sd = 0; sd < 4; ++sd) oacc[sd][i] *= scale;
            Ps[w][m_local][c] = f2bf(p0);
            Ps[w][m_local][16 + c] = f2bf(p1);
        }
        v16bf pa = load_a(&Ps[w][0][0], 40);
#pragma unroll
        for (int sd = 0; sd < 4; ++sd) {
            v16bf bv = load_b(&Vt[sd * 16][0], 40);
            oacc[sd] = __builtin_amdgcn_wmma_f32_16x16x32_bf16(
                false, pa, false, bv, (short)0, oacc[sd], false, false);
        }
    }
#pragma unroll
    for (int i = 0; i < 8; ++i) {
        const int qq = qb * 64 + w * 16 + i + 8 * hh;
        float invl = 1.0f / lrow[i];
#pragma unroll
        for (int sd = 0; sd < 4; ++sd)
            attb[headBase + (size_t)qq * Dn + sd * 16 + c] = f2bf(oacc[sd][i] * invl);
    }
}

// ============== Kernel 4: output projection + bias ===================
// out[m][n] = sum_k att[m][k]*Wproj[n][k] + bias[n]; att bf16 in [B][H][T][D].
// A tiles are 2D in memory: 32 contiguous d, row stride 64 over t -> TDM.
__global__ __launch_bounds__(256) void out_proj(const __bf16* __restrict__ attb,
                                                const __bf16* __restrict__ Wb,
                                                const float* __restrict__ bias,
                                                float* __restrict__ out) {
    __shared__ __bf16 As[2][128][LDT];
    __shared__ __bf16 Bs[2][128][LDT];
    const int tid = threadIdx.x;
    const int m0 = blockIdx.x * 128;
    const int n0 = blockIdx.y * 128;
    const int w = tid >> 5, lane = tid & 31;
    const int wm = (w & 3) * 32, wn = (w >> 2) * 64;
    const int bb = m0 >> 11, t0 = m0 & (Tn - 1);

    const tv8i g1a = tdm_g1_2d(32u, 128u, 64ull);
    const tv8i g1b = tdm_g1_2d(32u, 128u, 1024ull);
    v8f acc[2][4] = {};

    const __bf16* abase = attb + ((size_t)(bb * Hn)) * Tn * Dn + (size_t)t0 * Dn;
    if (w == 0) { // k0 = 0 -> h=0, d=0
        tdm_load_2d(lds_off(&As[0][0][0]), abase, g1a);
        tdm_load_2d(lds_off(&Bs[0][0][0]), Wb + (size_t)n0 * Cn, g1b);
    }
    for (int kt = 0; kt < 32; ++kt) {
        const int cur = kt & 1, nxt = cur ^ 1;
        __syncthreads();
        if (w == 0) {
            if (kt + 1 < 32) {
                int k1 = (kt + 1) * 32;
                int hd = k1 >> 6, d = k1 & 63;
                tdm_load_2d(lds_off(&As[nxt][0][0]),
                            abase + (size_t)hd * Tn * Dn + d, g1a);
                tdm_load_2d(lds_off(&Bs[nxt][0][0]), Wb + (size_t)n0 * Cn + k1, g1b);
                __builtin_amdgcn_s_wait_tensorcnt(2);
            } else {
                __builtin_amdgcn_s_wait_tensorcnt(0);
            }
        }
        __syncthreads();
        v16bf a0 = load_a(&As[cur][wm][0], LDT);
        v16bf a1 = load_a(&As[cur][wm + 16][0], LDT);
        v16bf b0 = load_b(&Bs[cur][wn][0], LDT);
        v16bf b1 = load_b(&Bs[cur][wn + 16][0], LDT);
        v16bf b2 = load_b(&Bs[cur][wn + 32][0], LDT);
        v16bf b3 = load_b(&Bs[cur][wn + 48][0], LDT);
        acc[0][0] = __builtin_amdgcn_wmma_f32_16x16x32_bf16(false, a0, false, b0, (short)0, acc[0][0], false, false);
        acc[0][1] = __builtin_amdgcn_wmma_f32_16x16x32_bf16(false, a0, false, b1, (short)0, acc[0][1], false, false);
        acc[0][2] = __builtin_amdgcn_wmma_f32_16x16x32_bf16(false, a0, false, b2, (short)0, acc[0][2], false, false);
        acc[0][3] = __builtin_amdgcn_wmma_f32_16x16x32_bf16(false, a0, false, b3, (short)0, acc[0][3], false, false);
        acc[1][0] = __builtin_amdgcn_wmma_f32_16x16x32_bf16(false, a1, false, b0, (short)0, acc[1][0], false, false);
        acc[1][1] = __builtin_amdgcn_wmma_f32_16x16x32_bf16(false, a1, false, b1, (short)0, acc[1][1], false, false);
        acc[1][2] = __builtin_amdgcn_wmma_f32_16x16x32_bf16(false, a1, false, b2, (short)0, acc[1][2], false, false);
        acc[1][3] = __builtin_amdgcn_wmma_f32_16x16x32_bf16(false, a1, false, b3, (short)0, acc[1][3], false, false);
    }
    const int hh = lane >> 4, c = lane & 15;
#pragma unroll
    for (int mi = 0; mi < 2; ++mi) {
#pragma unroll
        for (int s = 0; s < 4; ++s) {
            int n = n0 + wn + s * 16 + c;
            float bv = bias[n];
#pragma unroll
            for (int i = 0; i < 8; ++i) {
                int m = m0 + wm + mi * 16 + i + 8 * hh;
                out[(size_t)m * Cn + n] = acc[mi][s][i] + bv;
            }
        }
    }
}

// ============================ launcher ===============================
extern "C" void kernel_launch(void* const* d_in, const int* in_sizes, int n_in,
                              void* d_out, int out_size, void* d_ws, size_t ws_size,
                              hipStream_t stream) {
    const float* x    = (const float*)d_in[0];
    const float* Wqkv = (const float*)d_in[1];
    const float* Wprj = (const float*)d_in[2];
    const float* bprj = (const float*)d_in[3];
    const float* x0w  = (const float*)d_in[4];
    const float* x0b  = (const float*)d_in[5];
    const float* hcur = (const float*)d_in[6];
    (void)in_sizes; (void)n_in; (void)out_size; (void)ws_size;

    float* qp = (float*)d_ws;            // [B][H][T][D] f32
    float* kp = qp + (size_t)BHTD;
    float* vp = qp + (size_t)2 * BHTD;
    __bf16* xb     = (__bf16*)(vp + (size_t)BHTD);       // 8388608 bf16
    __bf16* wqkvb  = xb + (size_t)8388608;               // 3145728 bf16
    __bf16* wprojb = wqkvb + (size_t)3145728;            // 1048576 bf16
    __bf16* attb   = wprojb + (size_t)1048576;           // 8388608 bf16

    // 0) pre-convert GEMM operands to bf16 (enables TDM direct-to-LDS)
    conv_bf16<<<8388608 / 1024, 256, 0, stream>>>(x, xb);
    conv_bf16<<<3145728 / 1024, 256, 0, stream>>>(Wqkv, wqkvb);
    conv_bf16<<<1048576 / 1024, 256, 0, stream>>>(Wprj, wprojb);
    // 1) QKV GEMM, TDM-fed; epilogue scatters f32 q/k/v planes
    qkv_gemm<<<dim3(8192 / 128, 3072 / 128), 256, 0, stream>>>(xb, wqkvb, qp);
    // 2) RoPE on q and k in place
    rope_kernel<<<(Bn * Hn * Tn * 32) / 256, 256, 0, stream>>>(qp, kp);
    // 3) causal 3-geometry flash attention -> attb (bf16)
    attn_kernel<<<dim3(Tn / 64, Hn, Bn), 128, 0, stream>>>(qp, kp, vp, x0w, x0b, hcur, attb);
    // 4) output projection + bias -> d_out [B,T,C] f32
    out_proj<<<dim3(8192 / 128, 1024 / 128), 256, 0, stream>>>(attb, wprojb, bprj, (float*)d_out);
}
